// NiN_Conv2D_54116587929941
// MI455X (gfx1250) — matrix operand order repaired and checked by reading, not compiled, not run on hardware
//
#include <hip/hip_runtime.h>
#include <stddef.h>

typedef __attribute__((ext_vector_type(16))) _Float16 v16h;
typedef __attribute__((ext_vector_type(8)))  _Float16 v8h;
typedef __attribute__((ext_vector_type(8)))  float    v8f;

#define NPIX   32768      // 32*32*32 pixels
#define KP     160        // padded patch length (147 -> 160, 5 K-blocks of 32)
#define KREAL  147
#define NF     128        // filters
#define D0     32
#define D1     16

// workspace layout (bytes)
#define P_OFF   0
#define P_BYTES ((size_t)NPIX * KP * 2)                      // 10,485,760
#define W0F_OFF (P_OFF + P_BYTES)
#define W0F_ELEMS ((size_t)NF * 5 * 2 * 32 * 16)             // 655,360 halfs
#define W1F_OFF (W0F_OFF + W0F_ELEMS * 2)
#define W1F_ELEMS ((size_t)NF * 32 * 16)                     // 65,536 halfs

// ---------------------------------------------------------------------------
// Prep 1: patch matrix P[n][k] (f16, row-major, zero-padded) from x (B,H,W,C)
// k = (i*7 + j)*3 + c, SAME zero padding (pad 3 each side)
// ---------------------------------------------------------------------------
__global__ void build_patches(const float* __restrict__ x, _Float16* __restrict__ P) {
    int tid = blockIdx.x * blockDim.x + threadIdx.x;
    if (tid >= NPIX * KP) return;
    int n = tid / KP;
    int k = tid - n * KP;
    float v = 0.0f;
    if (k < KREAL) {
        int c = k % 3;
        int r = k / 3;
        int j = r % 7;
        int i = r / 7;
        int w = n & 31;
        int h = (n >> 5) & 31;
        int b = n >> 10;
        int hh = h + i - 3;
        int ww = w + j - 3;
        if (hh >= 0 && hh < 32 && ww >= 0 && ww < 32)
            v = x[(((size_t)b * 32 + hh) * 32 + ww) * 3 + c];
    }
    P[tid] = (_Float16)v;
}

// ---------------------------------------------------------------------------
// Prep 2: w0 (147,32,128) -> per-lane WMMA B fragments, f16, zero-padded K.
// Layout: [f][kb(5)][nt(2)][lane(32)][idx(16)].  B k order used here (contiguous, as in A loads):
//   lane L: N = L%16 + nt*16,  K(kb,idx) = kb*32 + (L/16)*16 + idx
// ---------------------------------------------------------------------------
__global__ void build_w0frag(const float* __restrict__ w0, _Float16* __restrict__ W0f) {
    int tid = blockIdx.x * blockDim.x + threadIdx.x;
    if (tid >= (int)W0F_ELEMS) return;
    int idx  = tid & 15;
    int t1   = tid >> 4;
    int lane = t1 & 31;
    int t2   = t1 >> 5;
    int nt   = t2 & 1;
    int t3   = t2 >> 1;
    int kb   = t3 % 5;
    int f    = t3 / 5;
    int grp  = lane >> 4;
    int K    = kb * 32 + grp * 16 + idx;
    int d    = nt * 16 + (lane & 15);
    float v = (K < KREAL) ? w0[((size_t)K * D0 + d) * NF + f] : 0.0f;
    W0f[tid] = (_Float16)v;
}

// ---------------------------------------------------------------------------
// Prep 3: w1 (32,16,128) -> B fragments [f][lane(32)][idx(16)], K=32 exactly.
// ---------------------------------------------------------------------------
__global__ void build_w1frag(const float* __restrict__ w1, _Float16* __restrict__ W1f) {
    int tid = blockIdx.x * blockDim.x + threadIdx.x;
    if (tid >= (int)W1F_ELEMS) return;
    int idx  = tid & 15;
    int lane = (tid >> 4) & 31;
    int f    = tid >> 9;
    int grp  = lane >> 4;
    int K    = grp * 16 + idx;         // d0 index 0..31
    int e    = lane & 15;              // d1 index
    W1f[tid] = (_Float16)w1[((size_t)K * D1 + e) * NF + f];
}

// ---------------------------------------------------------------------------
// Main kernel: grid = (F=128, 32 pixel chunks), block = 256 (8 waves).
// Each wave processes 8 tiles of 16 pixels:
//   layer0: 5 K-blocks x 2 n-tiles of v_wmma_f32_16x16x32_f16
//   layer1: 1 wmma (K = d0 = 32)
//   layer2: 16-wide dot via wave-private LDS
// ---------------------------------------------------------------------------
union AFrag { v16h v; struct { v8h lo, hi; } h; };

__global__ __launch_bounds__(256)
void nin_main(const _Float16* __restrict__ P,
              const _Float16* __restrict__ W0f,
              const _Float16* __restrict__ W1f,
              const float* __restrict__ b0,
              const float* __restrict__ b1,
              const float* __restrict__ w2,
              const float* __restrict__ b2,
              float* __restrict__ out) {
    __shared__ _Float16 h0s[8][16 * 32];   // per-wave 16x32 f16 tile
    __shared__ float    h1s[8][16 * 16];   // per-wave 16x16 f32 tile
    __shared__ float    w2s[16];

    const int f    = blockIdx.x;
    const int lane = threadIdx.x & 31;
    const int wave = threadIdx.x >> 5;
    const int grp  = lane >> 4;           // half-wave group
    const int ln   = lane & 15;

    if (threadIdx.x < 16) w2s[threadIdx.x] = w2[threadIdx.x * NF + f];
    __syncthreads();

    // Persistent B fragments for this filter f (L2-resident).
    v16h bw0[5][2];
#pragma unroll
    for (int kb = 0; kb < 5; ++kb)
#pragma unroll
        for (int nt = 0; nt < 2; ++nt)
            bw0[kb][nt] = *(const v16h*)(W0f +
                ((((size_t)f * 5 + kb) * 2 + nt) * 32 + lane) * 16);
    v16h bw1 = *(const v16h*)(W1f + ((size_t)f * 32 + lane) * 16);

    // Per-lane bias broadcast: C layout N = lane%16 (d index), all VGPRs share it.
    const float bias0a = b0[(size_t)ln * NF + f];
    const float bias0b = b0[(size_t)(16 + ln) * NF + f];
    const float bias1  = b1[(size_t)ln * NF + f];
    const float b2f    = b2[f];

    _Float16* h0 = h0s[wave];
    float*    h1 = h1s[wave];

    const int pixBase0 = blockIdx.y * 1024 + wave * 128;

    for (int t = 0; t < 8; ++t) {
        const int pixBase = pixBase0 + t * 16;
        const _Float16* prow = P + (size_t)(pixBase + ln) * KP;

        v8f acc0, acc1;
#pragma unroll
        for (int v = 0; v < 8; ++v) { acc0[v] = bias0a; acc1[v] = bias0b; }

        // layer 0: 10 WMMAs
#pragma unroll
        for (int kb = 0; kb < 5; ++kb) {
            AFrag a;
            a.h.lo = *(const v8h*)(prow + kb * 32 + grp * 16);        // K = grp*16 .. +7
            a.h.hi = *(const v8h*)(prow + kb * 32 + grp * 16 + 8);   // K = grp*16+8 ..
            acc0 = __builtin_amdgcn_wmma_f32_16x16x32_f16(
                false, a.v, false, bw0[kb][0], (short)0, acc0, false, false);
            acc1 = __builtin_amdgcn_wmma_f32_16x16x32_f16(
                false, a.v, false, bw0[kb][1], (short)0, acc1, false, false);
        }

        // relu + f16 + transpose (D layout -> A layout) through wave-private LDS
#pragma unroll
        for (int v = 0; v < 8; ++v) {
            const int m = v + 8 * grp;
            h0[m * 32 + ln]      = (_Float16)fmaxf(acc0[v], 0.0f);
            h0[m * 32 + 16 + ln] = (_Float16)fmaxf(acc1[v], 0.0f);
        }

        // layer 1: single WMMA, K = 32
        AFrag a1;
        a1.h.lo = *(const v8h*)(h0 + ln * 32 + grp * 16);
        a1.h.hi = *(const v8h*)(h0 + ln * 32 + grp * 16 + 8);
        v8f c1;
#pragma unroll
        for (int v = 0; v < 8; ++v) c1[v] = bias1;
        c1 = __builtin_amdgcn_wmma_f32_16x16x32_f16(
            false, a1.v, false, bw1, (short)0, c1, false, false);

#pragma unroll
        for (int v = 0; v < 8; ++v)
            h1[(v + 8 * grp) * 16 + ln] = fmaxf(c1[v], 0.0f);

        // layer 2: 16-wide dot per pixel (lanes 0..15), relu, store (pixel, f)
        if (lane < 16) {
            float s = b2f;
#pragma unroll
            for (int e = 0; e < 16; ++e) s += h1[lane * 16 + e] * w2s[e];
            out[(size_t)(pixBase + lane) * NF + f] = fmaxf(s, 0.0f);
        }
    }
}

// ---------------------------------------------------------------------------
extern "C" void kernel_launch(void* const* d_in, const int* in_sizes, int n_in,
                              void* d_out, int out_size, void* d_ws, size_t ws_size,
                              hipStream_t stream) {
    const float* x  = (const float*)d_in[0];
    const float* w0 = (const float*)d_in[1];
    const float* b0 = (const float*)d_in[2];
    const float* w1 = (const float*)d_in[3];
    const float* b1 = (const float*)d_in[4];
    const float* w2 = (const float*)d_in[5];
    const float* b2 = (const float*)d_in[6];
    float* out = (float*)d_out;

    char* ws = (char*)d_ws;
    _Float16* P   = (_Float16*)(ws + P_OFF);
    _Float16* W0f = (_Float16*)(ws + W0F_OFF);
    _Float16* W1f = (_Float16*)(ws + W1F_OFF);

    build_patches<<<(NPIX * KP + 255) / 256, 256, 0, stream>>>(x, P);
    build_w0frag<<<((int)W0F_ELEMS + 255) / 256, 256, 0, stream>>>(w0, W0f);
    build_w1frag<<<((int)W1F_ELEMS + 255) / 256, 256, 0, stream>>>(w1, W1f);

    dim3 grid(NF, 32);
    nin_main<<<grid, 256, 0, stream>>>(P, W0f, W1f, b0, b1, w2, b2, out);
}
